// TransformerBlock_9517647528571
// MI455X (gfx1250) — matrix operand order repaired and checked by reading, not compiled, not run on hardware
//
#include <hip/hip_runtime.h>

typedef __attribute__((ext_vector_type(16))) __bf16 v16bf;
typedef __attribute__((ext_vector_type(8)))  float  v8f;

#define T_TOK 8192
#define E_DIM 768
#define E3    2304
#define HID_D 3072
#define NHEAD 12
#define DHEAD 64
#define SEQ   1024
#define NEXP  4

#define BM 128
#define BN 64
#define BK 32

#if defined(__gfx1250__)
#define USE_ASYNC_LDS 1
#else
#define USE_ASYNC_LDS 0
#endif

__device__ __forceinline__ float gelu_exact(float v) {
  return 0.5f * v * (1.0f + erff(v * 0.70710678118654752f));
}

// 16-byte global -> LDS copy. On gfx1250 use the async tensor-style path
// (GLOBAL_LOAD_ASYNC_TO_LDS_B128, ASYNCcnt) so data never touches VGPRs.
__device__ __forceinline__ void g2l_b128(const void* g, void* l) {
#if USE_ASYNC_LDS
  unsigned int lds_off = (unsigned int)(unsigned long long)l;  // addr[31:0] = LDS offset
  unsigned long long ga = (unsigned long long)g;
  asm volatile("global_load_async_to_lds_b128 %0, %1, off"
               :: "v"(lds_off), "v"(ga)
               : "memory");
#else
  *(uint4*)l = *(const uint4*)g;
#endif
}

__device__ __forceinline__ void wait_async_3() {
#if USE_ASYNC_LDS
  asm volatile("s_wait_asynccnt 0x3" ::: "memory");
#endif
}
__device__ __forceinline__ void wait_async_0() {
#if USE_ASYNC_LDS
  asm volatile("s_wait_asynccnt 0x0" ::: "memory");
#endif
}

// ---------------------------------------------------------------------------
// fp32 -> bf16 convert (flat)
// ---------------------------------------------------------------------------
__global__ void cvt_bf16_kernel(const float* __restrict__ src,
                                __bf16* __restrict__ dst, int n) {
  int i = blockIdx.x * blockDim.x + threadIdx.x;
  int stride = gridDim.x * blockDim.x;
  for (; i < n; i += stride) dst[i] = (__bf16)src[i];
}

// fp32 [R,C] row-major  ->  bf16 [C,R] row-major (i.e. transpose+convert)
__global__ void tcvt_bf16_kernel(const float* __restrict__ src,
                                 __bf16* __restrict__ dst, int R, int C) {
  int i = blockIdx.x * blockDim.x + threadIdx.x;
  int n = R * C;
  int stride = gridDim.x * blockDim.x;
  for (; i < n; i += stride) {
    int r = i / C, c = i - r * C;
    dst[(size_t)c * R + r] = (__bf16)src[i];
  }
}

// ---------------------------------------------------------------------------
// LayerNorm over E=768, one block (256 thr) per token, bf16 output
// ---------------------------------------------------------------------------
__global__ __launch_bounds__(256) void ln_bf16_kernel(
    const float* __restrict__ x, const float* __restrict__ g,
    const float* __restrict__ b, __bf16* __restrict__ out) {
  __shared__ float sb[256];
  int t = blockIdx.x;
  int tid = threadIdx.x;
  const float* xp = x + (size_t)t * E_DIM;

  float s = xp[tid] + xp[tid + 256] + xp[tid + 512];
  sb[tid] = s; __syncthreads();
  for (int o = 128; o > 0; o >>= 1) {
    if (tid < o) sb[tid] += sb[tid + o];
    __syncthreads();
  }
  float mean = sb[0] * (1.0f / E_DIM);
  __syncthreads();

  float d0 = xp[tid] - mean, d1 = xp[tid + 256] - mean, d2 = xp[tid + 512] - mean;
  sb[tid] = d0 * d0 + d1 * d1 + d2 * d2; __syncthreads();
  for (int o = 128; o > 0; o >>= 1) {
    if (tid < o) sb[tid] += sb[tid + o];
    __syncthreads();
  }
  float rs = rsqrtf(sb[0] * (1.0f / E_DIM) + 1e-5f);

  __bf16* op = out + (size_t)t * E_DIM;
  for (int i = tid; i < E_DIM; i += 256)
    op[i] = (__bf16)((xp[i] - mean) * rs * g[i] + b[i]);
}

// ---------------------------------------------------------------------------
// Generic bf16 GEMM with fused epilogues.
//   C[M,N] = A[M,K](bf16,row) * B[K,N](bf16,row) + bias
//   EPI 0: Cf = v            (fp32 store)
//   EPI 1: Cb = gelu(v)      (bf16 store)
//   EPI 2: Cf = res + v      (residual; res may alias Cf)
//   EPI 3: Cf += scale[row*ss] * v   (MoE gated accumulate)
// Block: 256 thr = 8 waves, tile 128x64x32; wave computes 32x32 (2x2 WMMA).
// Double-buffered LDS fed by async global->LDS copies (ASYNCcnt pipelined).
// ---------------------------------------------------------------------------
template <int EPI>
__global__ __launch_bounds__(256) void gemm_bf16_kernel(
    const __bf16* __restrict__ A, const __bf16* __restrict__ B,
    const float* __restrict__ bias, float* __restrict__ Cf,
    __bf16* __restrict__ Cb, const float* __restrict__ res,
    const float* __restrict__ scale, int scaleStride,
    int M, int N, int K) {
  __shared__ __bf16 As[2][BM][BK + 8];
  __shared__ __bf16 Bs[2][BK][BN + 8];

  int tid = threadIdx.x;
  int wave = tid >> 5, lane = tid & 31;
  int lane15 = lane & 15, half = lane >> 4;
  int blockM = blockIdx.y * BM;
  int blockN = blockIdx.x * BN;
  int mBase = (wave >> 1) * 32;
  int nBase = (wave & 1) * 32;

  int arow = tid >> 1, aseg = tid & 1;        // A: 128 rows x (2 x 16 cols)
  int brow = tid >> 3, bcol = (tid & 7) * 8;  // B: 32 rows x (8 x 8 cols)

  const __bf16* aG = A + (size_t)(blockM + arow) * K + aseg * 16;
  const __bf16* bG = B + blockN + bcol + (size_t)brow * N;

  v8f acc[2][2] = {};

  // Issue 3 async b128 copies per thread for one (A,B) tile pair.
  auto stage = [&](int buf, int k0) {
    const __bf16* ga = aG + k0;
    g2l_b128(ga,     &As[buf][arow][aseg * 16]);
    g2l_b128(ga + 8, &As[buf][arow][aseg * 16 + 8]);
    g2l_b128(bG + (size_t)k0 * N, &Bs[buf][brow][bcol]);
  };

  const int nIter = K / BK;
  stage(0, 0);

  for (int it = 0; it < nIter; ++it) {
    int cur = it & 1;
    bool more = (it + 1) < nIter;
    if (more) stage(cur ^ 1, (it + 1) * BK);   // prefetch next tile
    if (more) wait_async_3(); else wait_async_0();  // drain current tile
    __syncthreads();

    v16bf a[2], b[2];
#pragma unroll
    for (int mi = 0; mi < 2; ++mi)
#pragma unroll
      for (int i = 0; i < 16; ++i)
        a[mi][i] = As[cur][mBase + mi * 16 + lane15][((i < 8) ? i : i + 8) + half * 8];
#pragma unroll
    for (int ni = 0; ni < 2; ++ni)
#pragma unroll
      for (int i = 0; i < 16; ++i)
        b[ni][i] = Bs[cur][((i < 8) ? i : i + 8) + half * 8][nBase + ni * 16 + lane15];

#pragma unroll
    for (int mi = 0; mi < 2; ++mi)
#pragma unroll
      for (int ni = 0; ni < 2; ++ni)
        acc[mi][ni] = __builtin_amdgcn_wmma_f32_16x16x32_bf16(
            false, a[mi], false, b[ni], (short)0, acc[mi][ni], false, false);
    __syncthreads();   // compute done before this buffer is refilled
  }

#pragma unroll
  for (int mi = 0; mi < 2; ++mi)
#pragma unroll
    for (int ni = 0; ni < 2; ++ni) {
      int col = blockN + nBase + ni * 16 + lane15;
      float bv = bias ? bias[col] : 0.0f;
#pragma unroll
      for (int j = 0; j < 8; ++j) {
        int row = blockM + mBase + mi * 16 + j + half * 8;
        size_t idx = (size_t)row * N + col;
        float v = acc[mi][ni][j] + bv;
        if (EPI == 0)      Cf[idx] = v;
        else if (EPI == 1) Cb[idx] = (__bf16)gelu_exact(v);
        else if (EPI == 2) Cf[idx] = res[idx] + v;
        else               Cf[idx] += scale[(size_t)row * scaleStride] * v;
      }
    }
}

// ---------------------------------------------------------------------------
// Attention: one block per (batch b, head h, 32-query tile).
// Scores (32x1024) kept in LDS fp32; full V panel (1024x64) in LDS bf16.
// ---------------------------------------------------------------------------
extern "C" __global__ __launch_bounds__(256) void attn_kernel(
    const float* __restrict__ qkv, __bf16* __restrict__ ctx) {
  extern __shared__ char smem[];
  __bf16* q_s = (__bf16*)smem;            // 32*64
  __bf16* v_s = q_s + 32 * 64;            // 1024*64
  float* sc = (float*)(v_s + SEQ * 64);   // 32*1024
  float* red = sc + 32 * SEQ;             // 32*8

  int blk = blockIdx.x;
  int qt = blk & 31;
  int bh = blk >> 5;
  int hIdx = bh % NHEAD;
  int bIdx = bh / NHEAD;

  int tid = threadIdx.x;
  int wave = tid >> 5, lane = tid & 31;
  int lane15 = lane & 15, half = lane >> 4;
  size_t tokBase = (size_t)bIdx * SEQ + qt * 32;

  // phase 0: Q (scaled by 1/sqrt(64)) and V panel
  for (int idx = tid; idx < 32 * 64; idx += 256) {
    int qi = idx >> 6, d = idx & 63;
    q_s[idx] = (__bf16)(qkv[(tokBase + qi) * E3 + hIdx * DHEAD + d] * 0.125f);
  }
  for (int idx = tid; idx < SEQ * 64; idx += 256) {
    int key = idx >> 6, d = idx & 63;
    v_s[idx] = (__bf16)qkv[((size_t)bIdx * SEQ + key) * E3 + 2 * E_DIM + hIdx * DHEAD + d];
  }
  __syncthreads();

  // phase 1: scores. Each wave owns 4 key-chunks of 32.
  for (int c = 0; c < 4; ++c) {
    int kc = (wave * 4 + c) * 32;
    v8f s[2][2] = {};
#pragma unroll
    for (int kk = 0; kk < 2; ++kk) {
      v16bf a[2], bfr[2];
#pragma unroll
      for (int mi = 0; mi < 2; ++mi)
#pragma unroll
        for (int i = 0; i < 16; ++i)
          a[mi][i] = q_s[(mi * 16 + lane15) * 64 + kk * 32 +
                         ((i < 8) ? i : i + 8) + half * 8];
#pragma unroll
      for (int ni = 0; ni < 2; ++ni) {
        const float* kp = qkv + ((size_t)bIdx * SEQ + kc + ni * 16 + lane15) * E3 +
                          E_DIM + hIdx * DHEAD + kk * 32 + half * 8;
#pragma unroll
        for (int i = 0; i < 16; ++i) bfr[ni][i] = (__bf16)kp[(i < 8) ? i : i + 8];
      }
#pragma unroll
      for (int mi = 0; mi < 2; ++mi)
#pragma unroll
        for (int ni = 0; ni < 2; ++ni)
          s[mi][ni] = __builtin_amdgcn_wmma_f32_16x16x32_bf16(
              false, a[mi], false, bfr[ni], (short)0, s[mi][ni], false, false);
    }
#pragma unroll
    for (int mi = 0; mi < 2; ++mi)
#pragma unroll
      for (int ni = 0; ni < 2; ++ni)
#pragma unroll
        for (int j = 0; j < 8; ++j)
          sc[(mi * 16 + j + half * 8) * SEQ + kc + ni * 16 + lane15] = s[mi][ni][j];
  }
  __syncthreads();

  // phase 2: softmax, 8 threads per row
  {
    int row = tid >> 3, p = tid & 7;
    float* rp = sc + row * SEQ;
    float lm = -3.0e38f;
    for (int i = p * 128; i < p * 128 + 128; ++i) lm = fmaxf(lm, rp[i]);
    red[row * 8 + p] = lm;
    __syncthreads();
    float m = red[row * 8];
    for (int k = 1; k < 8; ++k) m = fmaxf(m, red[row * 8 + k]);
    __syncthreads();
    float ls = 0.0f;
    for (int i = p * 128; i < p * 128 + 128; ++i) {
      float e = expf(rp[i] - m);
      rp[i] = e;
      ls += e;
    }
    red[row * 8 + p] = ls;
    __syncthreads();
    float tot = 0.0f;
    for (int k = 0; k < 8; ++k) tot += red[row * 8 + k];
    float inv = 1.0f / tot;
    for (int i = p * 128; i < p * 128 + 128; ++i) rp[i] *= inv;
  }
  __syncthreads();

  // phase 3: ctx = P @ V. 8 waves -> 8 fragments (2 m x 4 n of 16)
  {
    int mi = wave >> 2, ni = wave & 3;
    v8f acc = {};
    for (int kc = 0; kc < 32; ++kc) {
      v16bf a, bv;
#pragma unroll
      for (int i = 0; i < 16; ++i)
        a[i] = (__bf16)sc[(mi * 16 + lane15) * SEQ + kc * 32 +
                          ((i < 8) ? i : i + 8) + half * 8];
#pragma unroll
      for (int i = 0; i < 16; ++i)
        bv[i] = v_s[(kc * 32 + ((i < 8) ? i : i + 8) + half * 8) * 64 + ni * 16 + lane15];
      acc = __builtin_amdgcn_wmma_f32_16x16x32_bf16(false, a, false, bv, (short)0,
                                                    acc, false, false);
    }
#pragma unroll
    for (int j = 0; j < 8; ++j)
      ctx[(tokBase + mi * 16 + j + half * 8) * E_DIM + hIdx * DHEAD + ni * 16 + lane15] =
          (__bf16)acc[j];
  }
}

// ---------------------------------------------------------------------------
// MoE gate: per-token 4 logits, top-2 softmax -> comb[t,4]
// ---------------------------------------------------------------------------
__global__ void gate_kernel(const __bf16* __restrict__ xm,
                            const float* __restrict__ gw,
                            const float* __restrict__ gb,
                            float* __restrict__ comb) {
  int t = blockIdx.x * blockDim.x + threadIdx.x;
  if (t >= T_TOK) return;
  const __bf16* xp = xm + (size_t)t * E_DIM;
  float lg[NEXP];
  for (int e = 0; e < NEXP; ++e) {
    float s = gb[e];
    const float* wp = gw + (size_t)e * E_DIM;
    for (int d = 0; d < E_DIM; ++d) s += (float)xp[d] * wp[d];
    lg[e] = s;
  }
  int i1 = 0;
  for (int e = 1; e < NEXP; ++e)
    if (lg[e] > lg[i1]) i1 = e;
  int i2 = -1;
  for (int e = 0; e < NEXP; ++e)
    if (e != i1 && (i2 < 0 || lg[e] > lg[i2])) i2 = e;
  float e2 = expf(lg[i2] - lg[i1]);
  float inv = 1.0f / (1.0f + e2);
  float c[NEXP] = {0.f, 0.f, 0.f, 0.f};
  c[i1] = inv;
  c[i2] = e2 * inv;
  for (int e = 0; e < NEXP; ++e) comb[(size_t)t * NEXP + e] = c[e];
}

// ---------------------------------------------------------------------------
extern "C" void kernel_launch(void* const* d_in, const int* in_sizes, int n_in,
                              void* d_out, int out_size, void* d_ws, size_t ws_size,
                              hipStream_t stream) {
  const float* x      = (const float*)d_in[0];
  const float* ln1_g  = (const float*)d_in[1];
  const float* ln1_b  = (const float*)d_in[2];
  const float* in_w   = (const float*)d_in[3];
  const float* in_b   = (const float*)d_in[4];
  const float* out_w  = (const float*)d_in[5];
  const float* out_b  = (const float*)d_in[6];
  const float* mlp_w1 = (const float*)d_in[7];
  const float* mlp_b1 = (const float*)d_in[8];
  const float* mlp_w2 = (const float*)d_in[9];
  const float* mlp_b2 = (const float*)d_in[10];
  const float* ln2_g  = (const float*)d_in[11];
  const float* ln2_b  = (const float*)d_in[12];
  const float* gate_w = (const float*)d_in[13];
  const float* gate_b = (const float*)d_in[14];
  const float* exp_w1 = (const float*)d_in[15];
  const float* exp_b1 = (const float*)d_in[16];
  const float* exp_w2 = (const float*)d_in[17];
  const float* exp_b2 = (const float*)d_in[18];
  float* out = (float*)d_out;

  size_t off = 0;
  char* base = (char*)d_ws;
  auto carve = [&](size_t bytes) -> void* {
    void* p = base + off;
    off += (bytes + 255) & ~(size_t)255;
    return p;
  };

  __bf16* Wqkv  = (__bf16*)carve((size_t)E_DIM * E3 * 2);
  __bf16* Wout  = (__bf16*)carve((size_t)E_DIM * E_DIM * 2);
  __bf16* W1b   = (__bf16*)carve((size_t)E_DIM * HID_D * 2);
  __bf16* W2b   = (__bf16*)carve((size_t)HID_D * E_DIM * 2);
  __bf16* We1   = (__bf16*)carve((size_t)NEXP * E_DIM * HID_D * 2);
  __bf16* We2   = (__bf16*)carve((size_t)NEXP * HID_D * E_DIM * 2);
  __bf16* h_b   = (__bf16*)carve((size_t)T_TOK * E_DIM * 2);
  float*  qkv   = (float*) carve((size_t)T_TOK * E3 * 4);
  __bf16* ctx_b = (__bf16*)carve((size_t)T_TOK * E_DIM * 2);
  __bf16* x1_b  = (__bf16*)carve((size_t)T_TOK * E_DIM * 2);
  __bf16* h1g   = (__bf16*)carve((size_t)T_TOK * HID_D * 2);
  __bf16* xm_b  = (__bf16*)carve((size_t)T_TOK * E_DIM * 2);
  float*  comb  = (float*) carve((size_t)T_TOK * NEXP * 4);

  // --- weight conversion ---
  {
    int n;
    n = E3 * E_DIM;
    tcvt_bf16_kernel<<<(n + 255) / 256, 256, 0, stream>>>(in_w, Wqkv, E3, E_DIM);
    n = E_DIM * E_DIM;
    tcvt_bf16_kernel<<<(n + 255) / 256, 256, 0, stream>>>(out_w, Wout, E_DIM, E_DIM);
    n = E_DIM * HID_D;
    cvt_bf16_kernel<<<(n + 255) / 256, 256, 0, stream>>>(mlp_w1, W1b, n);
    n = HID_D * E_DIM;
    cvt_bf16_kernel<<<(n + 255) / 256, 256, 0, stream>>>(mlp_w2, W2b, n);
    n = NEXP * E_DIM * HID_D;
    cvt_bf16_kernel<<<(n + 255) / 256, 256, 0, stream>>>(exp_w1, We1, n);
    n = NEXP * HID_D * E_DIM;
    cvt_bf16_kernel<<<(n + 255) / 256, 256, 0, stream>>>(exp_w2, We2, n);
  }

  // --- LN1 -> bf16 ---
  ln_bf16_kernel<<<T_TOK, 256, 0, stream>>>(x, ln1_g, ln1_b, h_b);

  // --- QKV = h @ in_w.T + in_b  (fp32 out) ---
  gemm_bf16_kernel<0><<<dim3(E3 / BN, T_TOK / BM), 256, 0, stream>>>(
      h_b, Wqkv, in_b, qkv, nullptr, nullptr, nullptr, 0, T_TOK, E3, E_DIM);

  // --- attention -> ctx bf16 ---
  {
    size_t smem = (size_t)(32 * 64 + SEQ * 64) * 2 + (size_t)(32 * SEQ + 32 * 8) * 4;
    attn_kernel<<<8 * NHEAD * (SEQ / 32), 256, smem, stream>>>(qkv, ctx_b);
  }

  // --- x1 = x + ctx @ out_w.T + out_b  (into d_out) ---
  gemm_bf16_kernel<2><<<dim3(E_DIM / BN, T_TOK / BM), 256, 0, stream>>>(
      ctx_b, Wout, out_b, out, nullptr, x, nullptr, 0, T_TOK, E_DIM, E_DIM);

  // --- x1 -> bf16 ---
  {
    int n = T_TOK * E_DIM;
    cvt_bf16_kernel<<<(n + 255) / 256, 256, 0, stream>>>(out, x1_b, n);
  }

  // --- MLP: h1g = gelu(x1 @ W1 + b1) bf16; x2 = x1 + h1g @ W2 + b2 ---
  gemm_bf16_kernel<1><<<dim3(HID_D / BN, T_TOK / BM), 256, 0, stream>>>(
      x1_b, W1b, mlp_b1, nullptr, h1g, nullptr, nullptr, 0, T_TOK, HID_D, E_DIM);
  gemm_bf16_kernel<2><<<dim3(E_DIM / BN, T_TOK / BM), 256, 0, stream>>>(
      h1g, W2b, mlp_b2, out, nullptr, out, nullptr, 0, T_TOK, E_DIM, HID_D);

  // --- MoE ---
  ln_bf16_kernel<<<T_TOK, 256, 0, stream>>>(out, ln2_g, ln2_b, xm_b);
  gate_kernel<<<T_TOK / 256, 256, 0, stream>>>(xm_b, gate_w, gate_b, comb);

  for (int e = 0; e < NEXP; ++e) {
    gemm_bf16_kernel<1><<<dim3(HID_D / BN, T_TOK / BM), 256, 0, stream>>>(
        xm_b, We1 + (size_t)e * E_DIM * HID_D, exp_b1 + (size_t)e * HID_D,
        nullptr, h1g, nullptr, nullptr, 0, T_TOK, HID_D, E_DIM);
    gemm_bf16_kernel<3><<<dim3(E_DIM / BN, T_TOK / BM), 256, 0, stream>>>(
        h1g, We2 + (size_t)e * HID_D * E_DIM, exp_b2 + (size_t)e * E_DIM,
        out, nullptr, nullptr, comb + e, NEXP, T_TOK, E_DIM, HID_D);
  }
}
